// CrossAttention_50843822850167
// MI455X (gfx1250) — matrix operand order repaired and checked by reading, not compiled, not run on hardware
//
#include <hip/hip_runtime.h>
#include <math.h>

// Problem constants (from the reference)
#define BB 2
#define HH 16
#define TT 2048
#define SS 2048
#define DD 2048
#define HDIM 128
#define SCALE_F 0.08838834764831845f  // 1/sqrt(128)

typedef __bf16 bf16;
typedef __bf16 v16bf __attribute__((ext_vector_type(16)));
typedef __bf16 v8bf  __attribute__((ext_vector_type(8)));
typedef __bf16 v4bf  __attribute__((ext_vector_type(4)));
typedef __bf16 v2bf  __attribute__((ext_vector_type(2)));
typedef float  v8f   __attribute__((ext_vector_type(8)));
typedef unsigned int u32x4 __attribute__((ext_vector_type(4)));
typedef int i32x8 __attribute__((ext_vector_type(8)));
typedef int i32x4 __attribute__((ext_vector_type(4)));

#if defined(__has_builtin)
#if __has_builtin(__builtin_amdgcn_tensor_load_to_lds) && \
    __has_builtin(__builtin_amdgcn_s_wait_tensorcnt)
#define USE_TDM 1
#endif
#endif
#ifndef USE_TDM
#define USE_TDM 0
#endif

static __device__ __forceinline__ v8f wmma_bf16f32(v16bf a, v16bf b, v8f c) {
  // D = A(16x32 bf16) * B(32x16 bf16) + C(16x16 f32)
  return __builtin_amdgcn_wmma_f32_16x16x32_bf16(false, a, false, b, (short)0, c,
                                                 false, false);
}

// ---------------------------------------------------------------------------
// fp32 -> bf16 conversion (vectorized 4-wide)
// ---------------------------------------------------------------------------
__global__ __launch_bounds__(256) void cvt_f32_bf16(const float* __restrict__ in,
                                                    bf16* __restrict__ out, int n) {
  int i = (blockIdx.x * blockDim.x + threadIdx.x) * 4;
  if (i + 3 < n) {
    float4 f = *(const float4*)(in + i);
    v4bf o;
    o[0] = (bf16)f.x; o[1] = (bf16)f.y; o[2] = (bf16)f.z; o[3] = (bf16)f.w;
    *(v4bf*)(out + i) = o;
  }
}

// ---------------------------------------------------------------------------
// GEMM: C[M,N] = A[M,K] * W[N,K]^T  (bf16 inputs, fp32 accumulate)
// Block = 256 threads = 8 waves, tile 128(M) x 128(N), K-step 32.
// Waves: 4 along M (32 rows each) x 2 along N (64 cols each).
// Tile staging via Tensor Data Mover (double-buffered LDS), one issuing wave.
// ---------------------------------------------------------------------------
#define LDSP 48  // padded LDS row stride (bf16): 32 data + 16 pad = 96B rows

#if USE_TDM
// Build D# per CDNA5 ISA and issue TENSOR_LOAD_TO_LDS.
// 2D tile: tile_dim0=32 elems (64B rows), tile_dim1=128 rows, data_size=2B.
// LDS padding: pad_interval=3 (every 16 DWORDs) + pad_amount=7 (8 DWORDs)
// -> 96B LDS row stride, matching LDSP.
static __device__ __forceinline__ void tdm_load_tile(const bf16* gaddr,
                                                     unsigned lds_off, int K) {
  unsigned long long ga = (unsigned long long)(uintptr_t)gaddr;
  u32x4 g0;
  g0[0] = 1u;                                   // count=1, user-mode descriptor
  g0[1] = lds_off;                              // lds_addr [63:32]
  g0[2] = (unsigned)(ga & 0xffffffffu);         // global_addr [95:64]
  g0[3] = (unsigned)((ga >> 32) & 0x01ffffffu)  // global_addr [120:96]
          | 0x80000000u;                        // type=2 ("image") [127:126]
  i32x8 g1;
  g1[0] = (1 << 16)    // data_size = 2 bytes
        | (1 << 20)    // pad_enable
        | (3 << 22)    // pad_interval: 16 DWORDs (64B) between pads
        | (7 << 25);   // pad_amount: 8 DWORDs (32B)
  unsigned td0 = (unsigned)K;      // tensor_dim0 (elements per row)
  unsigned td1 = 0x00100000u;      // tensor_dim1 (rows; generous, tile in-bounds)
  g1[1] = (int)((td0 & 0xffffu) << 16);                       // barrier_addr=0 | td0.lo
  g1[2] = (int)(((td0 >> 16) & 0xffffu) | ((td1 & 0xffffu) << 16));
  g1[3] = (int)(((td1 >> 16) & 0xffffu) | (32u << 16));       // tile_dim0 = 32
  g1[4] = (int)128;                                           // tile_dim1=128, tile_dim2=0
  unsigned long long st0 = (unsigned long long)K;             // tensor_dim0_stride
  g1[5] = (int)(st0 & 0xffffffffu);
  g1[6] = (int)((st0 >> 32) & 0xffffu);                       // + dim1_stride.lo16 = 0
  g1[7] = 0;
  i32x4 z4 = {0, 0, 0, 0};          // groups 2/3 unused (tile_dim2..4 = 0)
  i32x8 z8 = {0, 0, 0, 0, 0, 0, 0, 0};
  // amdgpu-toolchain (clang-23) 6-arg form: (g0, g1, g2, g3, g4, cpol)
  __builtin_amdgcn_tensor_load_to_lds(g0, g1, z4, z4, z8, 0);
}
#endif

static __device__ __forceinline__ void gemm_frags_mma(const bf16* __restrict__ sAc,
                                                      const bf16* __restrict__ sBc,
                                                      int wm, int wn, int l16, int hi,
                                                      v8f (&acc)[2][4]) {
  // A-operand (16x32): lanes 0-15 hold K 0..15; lanes 16-31 K 16..31
  v16bf af[2];
#pragma unroll
  for (int mt = 0; mt < 2; ++mt) {
    const bf16* p = &sAc[(wm + mt * 16 + l16) * LDSP + hi * 16];
    ((v8bf*)&af[mt])[0] = *(const v8bf*)p;
    ((v8bf*)&af[mt])[1] = *(const v8bf*)(p + 8);
  }
  // B-operand (32x16): lanes 0-15 = K 0..15 of column l16; lanes 16-31 = K 16..31
  v16bf bf_[4];
#pragma unroll
  for (int nt = 0; nt < 4; ++nt) {
    const bf16* p = &sBc[(wn + nt * 16 + l16) * LDSP + hi * 16];
    ((v8bf*)&bf_[nt])[0] = *(const v8bf*)p;
    ((v8bf*)&bf_[nt])[1] = *(const v8bf*)(p + 8);
  }
#pragma unroll
  for (int mt = 0; mt < 2; ++mt)
#pragma unroll
    for (int nt = 0; nt < 4; ++nt)
      acc[mt][nt] = wmma_bf16f32(af[mt], bf_[nt], acc[mt][nt]);
}

template <bool OUT_BF16>
__global__ __launch_bounds__(256) void gemm_bf16_nt(const bf16* __restrict__ A,
                                                    const bf16* __restrict__ W,
                                                    void* __restrict__ Cout,
                                                    int M, int N, int K) {
  __shared__ bf16 sA[2][128 * LDSP];
  __shared__ bf16 sB[2][128 * LDSP];

  const int tid  = threadIdx.x;
  const int lane = tid & 31;
  const int wave = tid >> 5;
  const int l16  = lane & 15;
  const int hi   = lane >> 4;

  const int bm = blockIdx.x * 128;
  const int bn = blockIdx.y * 128;
  const int wm = (wave >> 1) * 32;
  const int wn = (wave & 1) * 64;

  v8f acc[2][4];
  for (int mt = 0; mt < 2; ++mt)
    for (int nt = 0; nt < 4; ++nt)
      for (int j = 0; j < 8; ++j) acc[mt][nt][j] = 0.0f;

#if USE_TDM
  const bool issuer = (wave == 0);
  const bf16* gA = A + (size_t)bm * K;
  const bf16* gW = W + (size_t)bn * K;
  if (issuer) {
    tdm_load_tile(gA, (unsigned)(uintptr_t)&sA[0][0], K);
    tdm_load_tile(gW, (unsigned)(uintptr_t)&sB[0][0], K);
  }
#pragma unroll 1
  for (int k0 = 0; k0 < K; k0 += 32) {
    const int cur = (k0 >> 5) & 1;
    if (issuer) __builtin_amdgcn_s_wait_tensorcnt(0);  // tile k0 resident
    __syncthreads();  // publish tile; also guarantees prev compute finished
    if (issuer && (k0 + 32) < K) {  // DMA next tile, overlapped with compute
      tdm_load_tile(gA + (k0 + 32), (unsigned)(uintptr_t)&sA[cur ^ 1][0], K);
      tdm_load_tile(gW + (k0 + 32), (unsigned)(uintptr_t)&sB[cur ^ 1][0], K);
    }
    gemm_frags_mma(sA[cur], sB[cur], wm, wn, l16, hi, acc);
  }
#else
  // Fallback: manual cooperative staging (thread t copies half a row).
  const int sr = tid >> 1;
  const int sc = (tid & 1) * 16;
#pragma unroll 1
  for (int k0 = 0; k0 < K; k0 += 32) {
    __syncthreads();
    const bf16* ga = A + (size_t)(bm + sr) * K + k0 + sc;
    *(v8bf*)&sA[0][sr * LDSP + sc]     = *(const v8bf*)ga;
    *(v8bf*)&sA[0][sr * LDSP + sc + 8] = *(const v8bf*)(ga + 8);
    const bf16* gw = W + (size_t)(bn + sr) * K + k0 + sc;
    *(v8bf*)&sB[0][sr * LDSP + sc]     = *(const v8bf*)gw;
    *(v8bf*)&sB[0][sr * LDSP + sc + 8] = *(const v8bf*)(gw + 8);
    __syncthreads();
    gemm_frags_mma(sA[0], sB[0], wm, wn, l16, hi, acc);
  }
#endif

  // Epilogue. C layout: VGPR j -> row j + 8*hi, col = l16 within each 16x16 tile.
#pragma unroll
  for (int mt = 0; mt < 2; ++mt)
#pragma unroll
    for (int nt = 0; nt < 4; ++nt)
#pragma unroll
      for (int j = 0; j < 8; ++j) {
        size_t r = (size_t)(bm + wm + mt * 16 + hi * 8 + j);
        size_t c = (size_t)(bn + wn + nt * 16 + l16);
        float v = acc[mt][nt][j];
        if (OUT_BF16)
          ((bf16*)Cout)[r * N + c] = (bf16)v;
        else
          ((float*)Cout)[r * N + c] = v;
      }
}

// ---------------------------------------------------------------------------
// RoPE + head split: X[B*L, D] (d = h*128+hd) -> Y[B,H,L,128], rotate pairs.
// ---------------------------------------------------------------------------
__global__ __launch_bounds__(256) void rope_split(const bf16* __restrict__ X,
                                                  const float* __restrict__ ctab,
                                                  const float* __restrict__ stab,
                                                  bf16* __restrict__ Y, int L) {
  int i = blockIdx.x * blockDim.x + threadIdx.x;  // (b,h,l,p), p fastest
  int p = i & 63;
  int l = (i >> 6) % L;
  int h = ((i >> 6) / L) % HH;
  int b = (i >> 6) / (L * HH);
  if (b >= BB) return;
  size_t src = ((size_t)(b * L + l)) * DD + h * HDIM + 2 * p;
  v2bf x = *(const v2bf*)(X + src);
  float x0 = (float)x[0], x1 = (float)x[1];
  float c = ctab[l * 64 + p], s = stab[l * 64 + p];
  v2bf y;
  y[0] = (bf16)(x0 * c - x1 * s);
  y[1] = (bf16)(x1 * c + x0 * s);
  size_t dst = (((size_t)(b * HH + h) * L) + l) * HDIM + 2 * p;
  *(v2bf*)(Y + dst) = y;
}

// ---------------------------------------------------------------------------
// V transpose: X[B*S, D] (d = h*128+hd) -> Vt[B,H,HD,S]
// ---------------------------------------------------------------------------
__global__ __launch_bounds__(256) void v_transpose(const bf16* __restrict__ X,
                                                   bf16* __restrict__ Y) {
  int i = blockIdx.x * blockDim.x + threadIdx.x;  // (b,h,hd,s), s fastest
  int s  = i & (SS - 1);
  int hd = (i >> 11) & (HDIM - 1);
  int h  = (i >> 18) & (HH - 1);
  int b  = i >> 22;
  if (b >= BB) return;
  Y[i] = X[((size_t)(b * SS + s)) * DD + h * HDIM + hd];
}

// ---------------------------------------------------------------------------
// Flash attention: 4 waves/block, each wave owns a 16-row Q tile of a 64-row
// block. Streams S in 64-column tiles: QK^T (16 WMMA) -> online softmax ->
// P@V (16 WMMA). K/V fragments come straight from global (L2-resident on
// MI455X's 192MB L2); next tile is prefetched one iteration ahead.
// ---------------------------------------------------------------------------
__global__ __launch_bounds__(128) void flash_attn(const bf16* __restrict__ Q,
                                                  const bf16* __restrict__ Kt,
                                                  const bf16* __restrict__ Vt,
                                                  bf16* __restrict__ ctx) {
  const int bh = blockIdx.x;      // b*H + h
  const int h  = bh & (HH - 1);
  const int b  = bh >> 4;
  const int q0 = blockIdx.y * 64;

  const int tid  = threadIdx.x;
  const int lane = tid & 31;
  const int wave = tid >> 5;
  const int l16  = lane & 15;
  const int hi   = lane >> 4;

  __shared__ bf16 sP[4][16 * 64];  // per-wave P tile, row-major [16][64]
  bf16* myP = sP[wave];

  // Resident Q fragments: 4 K-steps covering HD=128.
  const bf16* qrow = Q + ((size_t)bh * TT + q0 + wave * 16 + l16) * HDIM;
  v16bf qf[4];
#pragma unroll
  for (int kk = 0; kk < 4; ++kk) {
    const bf16* p = qrow + kk * 32 + hi * 16;
    ((v8bf*)&qf[kk])[0] = *(const v8bf*)p;
    ((v8bf*)&qf[kk])[1] = *(const v8bf*)(p + 8);
  }

  float m[8], lsum[8];
  v8f acc[8];
#pragma unroll
  for (int j = 0; j < 8; ++j) { m[j] = -INFINITY; lsum[j] = 0.0f; }
#pragma unroll
  for (int nt = 0; nt < 8; ++nt)
#pragma unroll
    for (int j = 0; j < 8; ++j) acc[nt][j] = 0.0f;

#pragma unroll 1
  for (int s0 = 0; s0 < SS; s0 += 64) {
    // Prefetch next s-tile of K and V (speculative, lands in L2/L0).
    if (s0 + 64 < SS) {
#pragma unroll
      for (int nt = 0; nt < 4; ++nt)
        __builtin_prefetch(Kt + ((size_t)bh * SS + s0 + 64 + nt * 16 + l16) * HDIM +
                               hi * 16, 0, 3);
#pragma unroll
      for (int nt = 0; nt < 8; ++nt)
        __builtin_prefetch(Vt + ((size_t)bh * HDIM + nt * 16 + l16) * SS + s0 + 64 +
                               hi * 16, 0, 3);
    }
    // ---- scores: 4 N-subtiles (s cols) x 4 K-steps (HD) ----
    v8f sc[4];
#pragma unroll
    for (int nt = 0; nt < 4; ++nt) {
#pragma unroll
      for (int j = 0; j < 8; ++j) sc[nt][j] = 0.0f;
#pragma unroll
      for (int kk = 0; kk < 4; ++kk) {
        const bf16* kp = Kt + ((size_t)bh * SS + s0 + nt * 16 + l16) * HDIM +
                         kk * 32 + hi * 16;
        v16bf kf;
        ((v8bf*)&kf)[0] = *(const v8bf*)kp;
        ((v8bf*)&kf)[1] = *(const v8bf*)(kp + 8);
        sc[nt] = wmma_bf16f32(qf[kk], kf, sc[nt]);
      }
    }
    // ---- online softmax (row reductions across the 16-lane N groups) ----
    float cand[8];
#pragma unroll
    for (int j = 0; j < 8; ++j) {
      float v = sc[0][j] * SCALE_F; sc[0][j] = v;
      float w;
      w = sc[1][j] * SCALE_F; sc[1][j] = w; v = fmaxf(v, w);
      w = sc[2][j] * SCALE_F; sc[2][j] = w; v = fmaxf(v, w);
      w = sc[3][j] * SCALE_F; sc[3][j] = w; v = fmaxf(v, w);
      cand[j] = v;
    }
#pragma unroll
    for (int mask = 1; mask < 16; mask <<= 1)
#pragma unroll
      for (int j = 0; j < 8; ++j)
        cand[j] = fmaxf(cand[j], __shfl_xor(cand[j], mask, 32));

    float corr[8], rs[8];
#pragma unroll
    for (int j = 0; j < 8; ++j) {
      float mn = fmaxf(m[j], cand[j]);
      corr[j] = __expf(m[j] - mn);
      m[j] = mn;
      rs[j] = 0.0f;
    }
    // P = exp(s - m) -> LDS (C-layout scatter to row-major 16x64)
#pragma unroll
    for (int nt = 0; nt < 4; ++nt)
#pragma unroll
      for (int j = 0; j < 8; ++j) {
        float pv = __expf(sc[nt][j] - m[j]);
        rs[j] += pv;
        myP[(j + hi * 8) * 64 + nt * 16 + l16] = (bf16)pv;
      }
#pragma unroll
    for (int mask = 1; mask < 16; mask <<= 1)
#pragma unroll
      for (int j = 0; j < 8; ++j) rs[j] += __shfl_xor(rs[j], mask, 32);
#pragma unroll
    for (int j = 0; j < 8; ++j) lsum[j] = lsum[j] * corr[j] + rs[j];
#pragma unroll
    for (int nt = 0; nt < 8; ++nt)
#pragma unroll
      for (int j = 0; j < 8; ++j) acc[nt][j] *= corr[j];

    // ---- P @ V: 2 K-steps (s) x 8 N-subtiles (HD) ----
#pragma unroll
    for (int ks = 0; ks < 2; ++ks) {
      v16bf pf;
      const bf16* pp = myP + l16 * 64 + ks * 32 + hi * 16;
      ((v8bf*)&pf)[0] = *(const v8bf*)pp;
      ((v8bf*)&pf)[1] = *(const v8bf*)(pp + 8);
#pragma unroll
      for (int nt = 0; nt < 8; ++nt) {
        const bf16* vp = Vt + ((size_t)bh * HDIM + nt * 16 + l16) * SS + s0 +
                         ks * 32 + hi * 16;
        v16bf vf;
        ((v8bf*)&vf)[0] = *(const v8bf*)vp;
        ((v8bf*)&vf)[1] = *(const v8bf*)(vp + 8);
        acc[nt] = wmma_bf16f32(pf, vf, acc[nt]);
      }
    }
  }

  // ---- normalize + store ctx (merge heads: d = h*128 + hd) ----
#pragma unroll
  for (int nt = 0; nt < 8; ++nt)
#pragma unroll
    for (int j = 0; j < 8; ++j) {
      int t = q0 + wave * 16 + hi * 8 + j;
      int d = h * HDIM + nt * 16 + l16;
      ctx[((size_t)b * TT + t) * DD + d] = (bf16)(acc[nt][j] / lsum[j]);
    }
}

// ---------------------------------------------------------------------------
// Host orchestration
// ---------------------------------------------------------------------------
extern "C" void kernel_launch(void* const* d_in, const int* in_sizes, int n_in,
                              void* d_out, int out_size, void* d_ws, size_t ws_size,
                              hipStream_t stream) {
  const float* hs   = (const float*)d_in[0];  // [B,T,D]
  const float* es   = (const float*)d_in[1];  // [B,S,D]
  const float* Wq   = (const float*)d_in[2];
  const float* Wk   = (const float*)d_in[3];
  const float* Wv   = (const float*)d_in[4];
  const float* Wo   = (const float*)d_in[5];
  const float* cosq = (const float*)d_in[6];  // [T,64]
  const float* sinq = (const float*)d_in[7];
  const float* cosk = (const float*)d_in[8];
  const float* sink = (const float*)d_in[9];
  float* out = (float*)d_out;

  const size_t MT  = (size_t)BB * TT;                 // 4096 rows
  const size_t XSZ = MT * DD * sizeof(bf16);          // 16 MB
  const size_t WSZ = (size_t)DD * DD * sizeof(bf16);  // 8 MB

  char* w = (char*)d_ws;
  bf16* Xh  = (bf16*)(w);
  bf16* Xe  = (bf16*)(w + XSZ);
  bf16* wq  = (bf16*)(w + 2 * XSZ);
  bf16* wk  = (bf16*)(w + 2 * XSZ + WSZ);
  bf16* wv  = (bf16*)(w + 2 * XSZ + 2 * WSZ);
  bf16* wo  = (bf16*)(w + 2 * XSZ + 3 * WSZ);
  bf16* Qt  = (bf16*)(w + 2 * XSZ + 4 * WSZ);  // Q proj (pre-RoPE)
  bf16* Ktm = (bf16*)(w + 3 * XSZ + 4 * WSZ);  // K proj (pre-RoPE)
  bf16* Vtm = (bf16*)(w + 4 * XSZ + 4 * WSZ);  // V proj
  bf16* Qr  = (bf16*)(w + 5 * XSZ + 4 * WSZ);  // Q [B,H,T,128]
  bf16* Kr  = (bf16*)(w + 6 * XSZ + 4 * WSZ);  // K [B,H,S,128]
  bf16* Vt  = (bf16*)(w + 7 * XSZ + 4 * WSZ);  // V^T [B,H,128,S]
  bf16* ctx = (bf16*)(w + 8 * XSZ + 4 * WSZ);  // ctx [B,T,D]

  const int nX = (int)(MT * DD);  // 8,388,608
  const int nW = DD * DD;         // 4,194,304

  // 1) fp32 -> bf16
  cvt_f32_bf16<<<nX / 1024, 256, 0, stream>>>(hs, Xh, nX);
  cvt_f32_bf16<<<nX / 1024, 256, 0, stream>>>(es, Xe, nX);
  cvt_f32_bf16<<<nW / 1024, 256, 0, stream>>>(Wq, wq, nW);
  cvt_f32_bf16<<<nW / 1024, 256, 0, stream>>>(Wk, wk, nW);
  cvt_f32_bf16<<<nW / 1024, 256, 0, stream>>>(Wv, wv, nW);
  cvt_f32_bf16<<<nW / 1024, 256, 0, stream>>>(Wo, wo, nW);

  // 2) projections: y = x @ W^T, bf16 out
  dim3 ggrid((unsigned)(MT / 128), (unsigned)(DD / 128));
  gemm_bf16_nt<true><<<ggrid, 256, 0, stream>>>(Xh, wq, Qt, (int)MT, DD, DD);
  gemm_bf16_nt<true><<<ggrid, 256, 0, stream>>>(Xe, wk, Ktm, (int)MT, DD, DD);
  gemm_bf16_nt<true><<<ggrid, 256, 0, stream>>>(Xe, wv, Vtm, (int)MT, DD, DD);

  // 3) RoPE + head split; V transpose
  const int npair = BB * HH * TT * 64;  // 4,194,304
  rope_split<<<npair / 256, 256, 0, stream>>>(Qt, cosq, sinq, Qr, TT);
  rope_split<<<npair / 256, 256, 0, stream>>>(Ktm, cosk, sink, Kr, SS);
  v_transpose<<<nX / 256, 256, 0, stream>>>(Vtm, Vt);

  // 4) flash attention -> ctx
  dim3 fgrid((unsigned)(BB * HH), (unsigned)(TT / 64));
  flash_attn<<<fgrid, 128, 0, stream>>>(Qr, Kr, Vt, ctx);

  // 5) output projection, fp32 out
  gemm_bf16_nt<false><<<ggrid, 256, 0, stream>>>(ctx, wo, out, (int)MT, DD, DD);
}
